// MultiHeadAttention_24979529793739
// MI455X (gfx1250) — compile-verified, matrix operand order checked by reading, not run on hardware
//
#include <hip/hip_runtime.h>
#include <stdint.h>

// ---------------------------------------------------------------------------
// MHA for MI455X (gfx1250, wave32). All matmuls via v_wmma_f32_16x16x32_bf16.
// ~34 GFLOP vs ~60MB of traffic (fits in 192MB L2) => matrix-engine bound.
// GEMM tiles DMA'd into padded LDS by the Tensor Data Mover with a 2-deep
// software pipeline (issue next stage, compute, s_wait_tensorcnt), mask
// pre-packed to bits, flash-attention online softmax mapped onto the WMMA
// C-fragment row layout.
// ---------------------------------------------------------------------------

#define SEQ     2048
#define EMB     1024
#define NHEAD   16
#define HDIM    64
#define KT      32          // GEMM K-stage
#define LDS_KT  40          // padded GEMM stage stride: 32 + 8 elems (80B row)
#define LDS_K   72          // padded stride for 64-wide attention P scratch

typedef __attribute__((ext_vector_type(16))) __bf16 v16bf;
typedef __attribute__((ext_vector_type(8)))  float  v8f;

union Frag16 { uint4 q[2]; v16bf v; };

static __device__ inline uint16_t f32_to_bf16_rne(float f) {
  union { float f; uint32_t u; } x; x.f = f;
  uint32_t u = x.u + 0x7FFFu + ((x.u >> 16) & 1u);
  return (uint16_t)(u >> 16);
}

// 16x32 bf16 WMMA A/B fragment from an LDS tile (row stride `ld` elements).
// ISA 16-bit A layout: lanes 0-15 -> rows R..R+15, K 0..7 & 16..23;
// lanes 16-31 -> same rows, K 8..15 & 24..31.
static __device__ inline Frag16 lds_load_frag(const uint16_t* lds, int R, int ld, int K0) {
  Frag16 f;
  const int lane = threadIdx.x & 31;
  const int row  = R + (lane & 15);
  const int hi8  = (lane >> 4) << 3;
  const uint16_t* p = lds + row * ld + K0 + hi8;
  f.q[0] = *(const uint4*)(p);
  f.q[1] = *(const uint4*)(p + 16);
  return f;
}

// Same fragment sourced straight from global memory (row stride ld elements).
static __device__ inline Frag16 g_load_frag(const uint16_t* g, int R, int ld, int K0) {
  Frag16 f;
  const int lane = threadIdx.x & 31;
  const int row  = R + (lane & 15);
  const int hi8  = (lane >> 4) << 3;
  const uint16_t* p = g + (size_t)row * ld + K0 + hi8;
  f.q[0] = *(const uint4*)(p);
  f.q[1] = *(const uint4*)(p + 16);
  return f;
}

static __device__ inline v8f wmma_bf16(const Frag16& a, const Frag16& b, v8f c) {
  return __builtin_amdgcn_wmma_f32_16x16x32_bf16(false, a.v, false, b.v,
                                                 (short)0, c, false, false);
}

// ---------------------------------------------------------------------------
// Tensor Data Mover: 2D tile (32 x 128 rows, bf16) global -> LDS, padding
// 4 DWORDs (16B) after every 16 DWORDs (64B row) => LDS_KT (40 elem) stride.
// 6-arg builtin form (clang-23 / therock-10.0 headers).
// ---------------------------------------------------------------------------
#if defined(__has_builtin)
#if __has_builtin(__builtin_amdgcn_tensor_load_to_lds)
#define HAVE_TDM 1
#endif
#endif

#ifdef HAVE_TDM
typedef __attribute__((ext_vector_type(4))) unsigned int v4u;
typedef __attribute__((ext_vector_type(8))) int v8i;
typedef __attribute__((ext_vector_type(4))) int v4i;

static __device__ inline void tdm_tile_2d(const uint16_t* lds_dst, const uint16_t* gsrc) {
  const uint64_t ga = (uint64_t)(uintptr_t)gsrc;
  const uint32_t la = (uint32_t)(uint64_t)(uintptr_t)lds_dst;  // addr[31:0] = LDS offset
  v4u g0;
  g0[0] = 1u;                                                  // count=1
  g0[1] = la;                                                  // lds_addr
  g0[2] = (uint32_t)ga;                                        // global_addr[31:0]
  g0[3] = (uint32_t)((ga >> 32) & 0x01FFFFFFu) | (2u << 30);   // ga[56:32] | type=2
  v8i g1;
  g1[0] = (int)((1u << 16)      // data_size = 2 bytes
              | (1u << 20)      // pad_enable
              | (3u << 22)      // pad_interval: 16 DWORDs (one 64B row)
              | (3u << 25));    // pad_amount:   4 DWORDs (16B)
  g1[1] = (int)((unsigned)KT << 16);            // tensor_dim0[15:0]
  g1[2] = (int)(128u << 16);                    // tensor_dim1[15:0] (=128 rows)
  g1[3] = (int)((unsigned)KT << 16);            // tile_dim0 = 32 elements
  g1[4] = 128;                                  // tile_dim1 = 128 rows
  g1[5] = (int)EMB;                             // tensor_dim0_stride = 1024 elements
  g1[6] = 0;
  g1[7] = 0;
  const v4i gz4 = {0, 0, 0, 0};
  const v8i gz8 = {0, 0, 0, 0, 0, 0, 0, 0};
  __builtin_amdgcn_tensor_load_to_lds(g0, g1, gz4, gz4, gz8, 0);
}
#endif

// ---------------------------------------------------------------------------
// Elementwise f32 -> bf16 (one-time conversion of q/k/v and weights).
// ---------------------------------------------------------------------------
__global__ __launch_bounds__(256)
void mha_cvt_bf16(const float* __restrict__ src, uint16_t* __restrict__ dst) {
  const int i = blockIdx.x * 256 + threadIdx.x;
  float4 x = ((const float4*)src)[i];
  ushort4 u;
  u.x = f32_to_bf16_rne(x.x); u.y = f32_to_bf16_rne(x.y);
  u.z = f32_to_bf16_rne(x.z); u.w = f32_to_bf16_rne(x.w);
  ((ushort4*)dst)[i] = u;
}

// Pack boolean mask bytes into bits: word (M, nw) bit j = mask[M][nw*32+j].
__global__ __launch_bounds__(256)
void mha_mask_pack(const unsigned char* __restrict__ mask, uint32_t* __restrict__ mb) {
  const int idx = blockIdx.x * 256 + threadIdx.x;      // word index, grid is exact
  const uint32_t* src = (const uint32_t*)(mask + (size_t)idx * 32);
  uint32_t bits = 0;
#pragma unroll
  for (int i = 0; i < 8; ++i) {
    const uint32_t w = src[i];
    bits |= (uint32_t)((w & 0x000000FFu) != 0) << (4 * i + 0);
    bits |= (uint32_t)((w & 0x0000FF00u) != 0) << (4 * i + 1);
    bits |= (uint32_t)((w & 0x00FF0000u) != 0) << (4 * i + 2);
    bits |= (uint32_t)((w & 0xFF000000u) != 0) << (4 * i + 3);
  }
  mb[idx] = bits;
}

// ---------------------------------------------------------------------------
// bf16 GEMM: P = A @ B^T, A [S,E] bf16, B [E,E] bf16. Double-buffered TDM
// pipeline: issue next stage's DMA, run 8 WMMAs, then s_wait_tensorcnt.
// Block 256 thr (8 waves) -> 128x128 tile; wave grid 2x4, wave = 64x32.
// MODE 0: bf16 [H][S][64]; MODE 1: bf16 [H][64][S]; MODE 2: f32 [S][E].
// ---------------------------------------------------------------------------
template <int MODE>
__global__ __launch_bounds__(256)
void mha_gemm_bf16(const uint16_t* __restrict__ A, const uint16_t* __restrict__ B,
                   void* __restrict__ dst) {
  __shared__ uint16_t lA[2][128 * LDS_KT];
  __shared__ uint16_t lB[2][128 * LDS_KT];
  const int m0 = blockIdx.y * 128;
  const int n0 = blockIdx.x * 128;
  const int w  = threadIdx.x >> 5;
  const int Rw = (w >> 2) * 64;
  const int Cw = (w & 3) * 32;
  const int lane  = threadIdx.x & 31;
  const int mhalf = (lane >> 4) << 3;
  const int nl    = lane & 15;
  const uint16_t* Abase = A + (size_t)m0 * EMB;
  const uint16_t* Bbase = B + (size_t)n0 * EMB;

  v8f acc[4][2];
  const v8f vzero = {0.f,0.f,0.f,0.f,0.f,0.f,0.f,0.f};
#pragma unroll
  for (int i = 0; i < 4; ++i)
#pragma unroll
    for (int j = 0; j < 2; ++j) acc[i][j] = vzero;

#ifdef HAVE_TDM
  if (threadIdx.x < 32) {               // wave 0 drives the DMA engine
    tdm_tile_2d(&lA[0][0], Abase);
    tdm_tile_2d(&lB[0][0], Bbase);
    __builtin_amdgcn_s_wait_tensorcnt(0);
  }
  __syncthreads();
#endif

  int buf = 0;
  for (int k0 = 0; k0 < EMB; k0 += KT) {
#ifdef HAVE_TDM
    if (k0 + KT < EMB && threadIdx.x < 32) {   // prefetch next stage into other buffer
      tdm_tile_2d(&lA[buf ^ 1][0], Abase + k0 + KT);
      tdm_tile_2d(&lB[buf ^ 1][0], Bbase + k0 + KT);
    }
#else
    __syncthreads();
    for (int c = threadIdx.x; c < 128 * 4; c += 256) {   // 4 x uint4 chunks per row
      const int row = c >> 2, c8 = (c & 3) * 8;
      *(uint4*)&lA[0][row * LDS_KT + c8] = *(const uint4*)(Abase + (size_t)row * EMB + k0 + c8);
      *(uint4*)&lB[0][row * LDS_KT + c8] = *(const uint4*)(Bbase + (size_t)row * EMB + k0 + c8);
    }
    __syncthreads();
#endif
    Frag16 a[4];
#pragma unroll
    for (int i = 0; i < 4; ++i) a[i] = lds_load_frag(&lA[buf][0], Rw + 16 * i, LDS_KT, 0);
#pragma unroll
    for (int j = 0; j < 2; ++j) {
      Frag16 b = lds_load_frag(&lB[buf][0], Cw + 16 * j, LDS_KT, 0);
#pragma unroll
      for (int i = 0; i < 4; ++i) acc[i][j] = wmma_bf16(a[i], b, acc[i][j]);
    }
#ifdef HAVE_TDM
    if (threadIdx.x < 32) __builtin_amdgcn_s_wait_tensorcnt(0);
    __syncthreads();                    // next buffer ready; old buffer reusable
    buf ^= 1;
#endif
  }

#pragma unroll
  for (int i = 0; i < 4; ++i)
#pragma unroll
    for (int j = 0; j < 2; ++j)
#pragma unroll
      for (int r = 0; r < 8; ++r) {
        const int M = m0 + Rw + 16 * i + r + mhalf;
        const int N = n0 + Cw + 16 * j + nl;
        if (MODE == 2) {
          ((float*)dst)[(size_t)M * EMB + N] = acc[i][j][r];
        } else {
          const int h = N >> 6, d = N & 63;
          const uint16_t bv = f32_to_bf16_rne(acc[i][j][r]);
          if (MODE == 0) ((uint16_t*)dst)[((size_t)h * SEQ + M) * HDIM + d] = bv;
          else           ((uint16_t*)dst)[((size_t)h * HDIM + d) * SEQ + M] = bv;
        }
      }
}

// ---------------------------------------------------------------------------
// Fallback projection GEMM from f32 inputs (converts per-tile in LDS stage).
// Used when workspace is too small for the one-time bf16 conversion buffers.
// ---------------------------------------------------------------------------
template <int MODE>   // 0: [H][S][64], 1: [H][64][S]
__global__ __launch_bounds__(256)
void mha_proj_gemm_f32(const float* __restrict__ X, const float* __restrict__ W,
                       uint16_t* __restrict__ dst) {
  __shared__ uint16_t lA[128 * LDS_K];
  __shared__ uint16_t lB[128 * LDS_K];
  const int m0 = blockIdx.y * 128;
  const int n0 = blockIdx.x * 128;
  const int w  = threadIdx.x >> 5;
  const int Rw = (w >> 2) * 64;
  const int Cw = (w & 3) * 32;
  const int lane  = threadIdx.x & 31;
  const int mhalf = (lane >> 4) << 3;
  const int nl    = lane & 15;

  v8f acc[4][2];
  const v8f vzero = {0.f,0.f,0.f,0.f,0.f,0.f,0.f,0.f};
#pragma unroll
  for (int i = 0; i < 4; ++i)
#pragma unroll
    for (int j = 0; j < 2; ++j) acc[i][j] = vzero;

  for (int k0 = 0; k0 < EMB; k0 += 64) {
    __syncthreads();
    for (int c = threadIdx.x; c < 128 * 16; c += 256) {
      const int row = c >> 4, c4 = (c & 15) * 4;
      float4 xa = *(const float4*)(X + (size_t)(m0 + row) * EMB + k0 + c4);
      float4 xb = *(const float4*)(W + (size_t)(n0 + row) * EMB + k0 + c4);
      ushort4 ua, ub;
      ua.x = f32_to_bf16_rne(xa.x); ua.y = f32_to_bf16_rne(xa.y);
      ua.z = f32_to_bf16_rne(xa.z); ua.w = f32_to_bf16_rne(xa.w);
      ub.x = f32_to_bf16_rne(xb.x); ub.y = f32_to_bf16_rne(xb.y);
      ub.z = f32_to_bf16_rne(xb.z); ub.w = f32_to_bf16_rne(xb.w);
      *(ushort4*)&lA[row * LDS_K + c4] = ua;
      *(ushort4*)&lB[row * LDS_K + c4] = ub;
    }
    __syncthreads();
#pragma unroll
    for (int kk = 0; kk < 64; kk += 32) {
      Frag16 a[4];
#pragma unroll
      for (int i = 0; i < 4; ++i) a[i] = lds_load_frag(lA, Rw + 16 * i, LDS_K, kk);
#pragma unroll
      for (int j = 0; j < 2; ++j) {
        Frag16 b = lds_load_frag(lB, Cw + 16 * j, LDS_K, kk);
#pragma unroll
        for (int i = 0; i < 4; ++i) acc[i][j] = wmma_bf16(a[i], b, acc[i][j]);
      }
    }
  }
#pragma unroll
  for (int i = 0; i < 4; ++i)
#pragma unroll
    for (int j = 0; j < 2; ++j)
#pragma unroll
      for (int r = 0; r < 8; ++r) {
        const int M = m0 + Rw + 16 * i + r + mhalf;
        const int N = n0 + Cw + 16 * j + nl;
        const int h = N >> 6, d = N & 63;
        const uint16_t bv = f32_to_bf16_rne(acc[i][j][r]);
        if (MODE == 0) dst[((size_t)h * SEQ + M) * HDIM + d] = bv;
        else           dst[((size_t)h * HDIM + d) * SEQ + M] = bv;
      }
}

// Fallback output projection: Oc(bf16) @ Wo^T(f32) -> f32 d_out.
__global__ __launch_bounds__(256)
void mha_out_gemm_f32w(const uint16_t* __restrict__ Xb, const float* __restrict__ W,
                       float* __restrict__ out) {
  __shared__ uint16_t lA[128 * LDS_K];
  __shared__ uint16_t lB[128 * LDS_K];
  const int m0 = blockIdx.y * 128;
  const int n0 = blockIdx.x * 128;
  const int w  = threadIdx.x >> 5;
  const int Rw = (w >> 2) * 64;
  const int Cw = (w & 3) * 32;
  const int lane  = threadIdx.x & 31;
  const int mhalf = (lane >> 4) << 3;
  const int nl    = lane & 15;

  v8f acc[4][2];
  const v8f vzero = {0.f,0.f,0.f,0.f,0.f,0.f,0.f,0.f};
#pragma unroll
  for (int i = 0; i < 4; ++i)
#pragma unroll
    for (int j = 0; j < 2; ++j) acc[i][j] = vzero;

  for (int k0 = 0; k0 < EMB; k0 += 64) {
    __syncthreads();
    for (int c = threadIdx.x; c < 128 * 8; c += 256) {
      const int row = c >> 3, c8 = (c & 7) * 8;
      *(uint4*)&lA[row * LDS_K + c8] =
          *(const uint4*)(Xb + (size_t)(m0 + row) * EMB + k0 + c8);
    }
    for (int c = threadIdx.x; c < 128 * 16; c += 256) {
      const int row = c >> 4, c4 = (c & 15) * 4;
      float4 xb = *(const float4*)(W + (size_t)(n0 + row) * EMB + k0 + c4);
      ushort4 ub;
      ub.x = f32_to_bf16_rne(xb.x); ub.y = f32_to_bf16_rne(xb.y);
      ub.z = f32_to_bf16_rne(xb.z); ub.w = f32_to_bf16_rne(xb.w);
      *(ushort4*)&lB[row * LDS_K + c4] = ub;
    }
    __syncthreads();
#pragma unroll
    for (int kk = 0; kk < 64; kk += 32) {
      Frag16 a[4];
#pragma unroll
      for (int i = 0; i < 4; ++i) a[i] = lds_load_frag(lA, Rw + 16 * i, LDS_K, kk);
#pragma unroll
      for (int j = 0; j < 2; ++j) {
        Frag16 b = lds_load_frag(lB, Cw + 16 * j, LDS_K, kk);
#pragma unroll
        for (int i = 0; i < 4; ++i) acc[i][j] = wmma_bf16(a[i], b, acc[i][j]);
      }
    }
  }
#pragma unroll
  for (int i = 0; i < 4; ++i)
#pragma unroll
    for (int j = 0; j < 2; ++j)
#pragma unroll
      for (int r = 0; r < 8; ++r) {
        const int M = m0 + Rw + 16 * i + r + mhalf;
        const int N = n0 + Cw + 16 * j + nl;
        out[(size_t)M * EMB + N] = acc[i][j][r];
      }
}

// ---------------------------------------------------------------------------
// Flash attention with bit-packed mask. One block per (128 query rows, head);
// wave owns 16 rows; online softmax over 64-key tiles; stats live per-lane in
// the C-fragment row mapping, reduced via __shfl_xor over 16-lane halves.
// ---------------------------------------------------------------------------
__global__ __launch_bounds__(256)
void mha_flash_attn(const uint16_t* __restrict__ Qh, const uint16_t* __restrict__ Kh,
                    const uint16_t* __restrict__ Vt, const uint32_t* __restrict__ maskb,
                    uint16_t* __restrict__ Oc) {
  __shared__ uint16_t pS[8 * 16 * LDS_K];   // per-wave P-transpose scratch
  const int h    = blockIdx.y;
  const int w    = threadIdx.x >> 5;
  const int lane = threadIdx.x & 31;
  const int R    = blockIdx.x * 128 + w * 16;
  const int mhalf = (lane >> 4) << 3;
  const int nl    = lane & 15;

  const uint16_t* Q = Qh + (size_t)h * SEQ * HDIM;
  const uint16_t* K = Kh + (size_t)h * SEQ * HDIM;
  const uint16_t* V = Vt + (size_t)h * HDIM * SEQ;
  uint16_t* myP = pS + w * 16 * LDS_K;

  Frag16 qa0 = g_load_frag(Q, R, HDIM, 0);
  Frag16 qa1 = g_load_frag(Q, R, HDIM, 32);

  const v8f vzero = {0.f,0.f,0.f,0.f,0.f,0.f,0.f,0.f};
  v8f o[4];
  float mrow[8], lrow[8];
#pragma unroll
  for (int j = 0; j < 4; ++j) o[j] = vzero;
#pragma unroll
  for (int r = 0; r < 8; ++r) { mrow[r] = -3.0e38f; lrow[r] = 0.f; }

  for (int n0 = 0; n0 < SEQ; n0 += 64) {
    if (n0 + 64 < SEQ) {                       // global_prefetch_b8 of next tiles
      __builtin_prefetch(K + (size_t)(n0 + 64) * HDIM, 0, 1);
      __builtin_prefetch(V + n0 + 64, 0, 1);
    }
    // scores 16x64 = Q (16x64) @ K^T tile
    v8f sc[4];
#pragma unroll
    for (int t = 0; t < 4; ++t) {
      sc[t] = vzero;
      Frag16 b0 = g_load_frag(K, n0 + 16 * t, HDIM, 0);
      Frag16 b1 = g_load_frag(K, n0 + 16 * t, HDIM, 32);
      sc[t] = wmma_bf16(qa0, b0, sc[t]);
      sc[t] = wmma_bf16(qa1, b1, sc[t]);
    }
    // scale + packed mask (bit set -> -1e30)
#pragma unroll
    for (int r = 0; r < 8; ++r) {
      const int M = R + r + mhalf;
      const uint2 wm = *(const uint2*)(maskb + (size_t)M * (SEQ / 32) + (n0 >> 5));
      const uint32_t bx = wm.x >> nl, by = wm.y >> nl;
      float s;
      s = sc[0][r] * 0.125f; if (bx & 1u)          s = -1.0e30f; sc[0][r] = s;
      s = sc[1][r] * 0.125f; if ((bx >> 16) & 1u)  s = -1.0e30f; sc[1][r] = s;
      s = sc[2][r] * 0.125f; if (by & 1u)          s = -1.0e30f; sc[2][r] = s;
      s = sc[3][r] * 0.125f; if ((by >> 16) & 1u)  s = -1.0e30f; sc[3][r] = s;
    }
    // online softmax per row slot
#pragma unroll
    for (int r = 0; r < 8; ++r) {
      float tmax = fmaxf(fmaxf(sc[0][r], sc[1][r]), fmaxf(sc[2][r], sc[3][r]));
#pragma unroll
      for (int d = 1; d < 16; d <<= 1) tmax = fmaxf(tmax, __shfl_xor(tmax, d, 32));
      const float mn = fmaxf(mrow[r], tmax);
      const float f  = __expf(mrow[r] - mn);
      mrow[r] = mn;
      float rs = 0.f;
#pragma unroll
      for (int t = 0; t < 4; ++t) {
        const float p = __expf(sc[t][r] - mn);
        sc[t][r] = p;
        rs += p;
      }
#pragma unroll
      for (int d = 1; d < 16; d <<= 1) rs += __shfl_xor(rs, d, 32);
      lrow[r] = lrow[r] * f + rs;
#pragma unroll
      for (int j = 0; j < 4; ++j) o[j][r] *= f;
    }
    // C-layout P -> row-major bf16 in LDS -> A-fragments (LDS in-order per wave)
#pragma unroll
    for (int t = 0; t < 4; ++t)
#pragma unroll
      for (int r = 0; r < 8; ++r)
        myP[(r + mhalf) * LDS_K + 16 * t + nl] = f32_to_bf16_rne(sc[t][r]);
    asm volatile("" ::: "memory");
    Frag16 pa0 = lds_load_frag(myP, 0, LDS_K, 0);
    Frag16 pa1 = lds_load_frag(myP, 0, LDS_K, 32);
    // O += P @ V ; V stored transposed [H][64][S] so B reads contiguous n
#pragma unroll
    for (int j = 0; j < 4; ++j) {
      Frag16 b0 = g_load_frag(V, 16 * j, SEQ, n0);
      Frag16 b1 = g_load_frag(V, 16 * j, SEQ, n0 + 32);
      o[j] = wmma_bf16(pa0, b0, o[j]);
      o[j] = wmma_bf16(pa1, b1, o[j]);
    }
  }
  // normalize and store concat-heads bf16 [S][E]
#pragma unroll
  for (int j = 0; j < 4; ++j)
#pragma unroll
    for (int r = 0; r < 8; ++r) {
      const int M = R + r + mhalf;
      const int e = h * HDIM + 16 * j + nl;
      Oc[(size_t)M * EMB + e] = f32_to_bf16_rne(o[j][r] / lrow[r]);
    }
}

// ---------------------------------------------------------------------------
extern "C" void kernel_launch(void* const* d_in, const int* in_sizes, int n_in,
                              void* d_out, int out_size, void* d_ws, size_t ws_size,
                              hipStream_t stream) {
  (void)in_sizes; (void)n_in; (void)out_size;
  const float* q    = (const float*)d_in[0];
  const float* k    = (const float*)d_in[1];
  const float* v    = (const float*)d_in[2];
  const unsigned char* mask = (const unsigned char*)d_in[3];  // jax bool = 1 byte
  const float* Wq   = (const float*)d_in[4];
  const float* Wk   = (const float*)d_in[5];
  const float* Wv   = (const float*)d_in[6];
  const float* Wo   = (const float*)d_in[7];

  const size_t S_E = (size_t)SEQ * EMB;   // 2M elements
  const size_t E_E = (size_t)EMB * EMB;   // 1M elements

  // workspace layout (bf16 elements unless noted):
  uint16_t* Qh = (uint16_t*)d_ws;                 //  4 MB
  uint16_t* Kh = Qh + S_E;                        //  4 MB
  uint16_t* Vt = Kh + S_E;                        //  4 MB
  uint16_t* Oc = Vt + S_E;                        //  4 MB
  uint32_t* maskb = (uint32_t*)(Oc + S_E);        //  512 KB (bit-packed mask)
  uint16_t* big = (uint16_t*)(maskb + (size_t)SEQ * (SEQ / 32));
  // big path adds: qb/kb/vb (12 MB) + 4 bf16 weights (8 MB) => ~37 MB total
  const size_t need_big = (4 * S_E + 3 * S_E + 4 * E_E) * 2 + (size_t)SEQ * (SEQ / 32) * 4;
  const bool use_big = ws_size >= need_big;

  mha_mask_pack<<<(SEQ * (SEQ / 32)) / 256, 256, 0, stream>>>(mask, maskb);

  const dim3 gproj(EMB / 128, SEQ / 128);   // 8 x 16

  if (use_big) {
    uint16_t* qb  = big;
    uint16_t* kb  = qb + S_E;
    uint16_t* vb  = kb + S_E;
    uint16_t* wqb = vb + S_E;
    uint16_t* wkb = wqb + E_E;
    uint16_t* wvb = wkb + E_E;
    uint16_t* wob = wvb + E_E;
    mha_cvt_bf16<<<(int)(S_E / 4 / 256), 256, 0, stream>>>(q,  qb);
    mha_cvt_bf16<<<(int)(S_E / 4 / 256), 256, 0, stream>>>(k,  kb);
    mha_cvt_bf16<<<(int)(S_E / 4 / 256), 256, 0, stream>>>(v,  vb);
    mha_cvt_bf16<<<(int)(E_E / 4 / 256), 256, 0, stream>>>(Wq, wqb);
    mha_cvt_bf16<<<(int)(E_E / 4 / 256), 256, 0, stream>>>(Wk, wkb);
    mha_cvt_bf16<<<(int)(E_E / 4 / 256), 256, 0, stream>>>(Wv, wvb);
    mha_cvt_bf16<<<(int)(E_E / 4 / 256), 256, 0, stream>>>(Wo, wob);
    mha_gemm_bf16<0><<<gproj, 256, 0, stream>>>(qb, wqb, Qh);
    mha_gemm_bf16<0><<<gproj, 256, 0, stream>>>(kb, wkb, Kh);
    mha_gemm_bf16<1><<<gproj, 256, 0, stream>>>(vb, wvb, Vt);
  } else {
    mha_proj_gemm_f32<0><<<gproj, 256, 0, stream>>>(q, Wq, Qh);
    mha_proj_gemm_f32<0><<<gproj, 256, 0, stream>>>(k, Wk, Kh);
    mha_proj_gemm_f32<1><<<gproj, 256, 0, stream>>>(v, Wv, Vt);
  }

  mha_flash_attn<<<dim3(SEQ / 128, NHEAD), 256, 0, stream>>>(Qh, Kh, Vt, maskb, Oc);

  if (use_big) {
    uint16_t* wob = big + 3 * S_E + 3 * E_E;
    mha_gemm_bf16<2><<<gproj, 256, 0, stream>>>(Oc, wob, (float*)d_out);
  } else {
    mha_out_gemm_f32w<<<gproj, 256, 0, stream>>>(Oc, Wo, (float*)d_out);
  }
}
